// BoidsRouterLoss_12936441495903
// MI455X (gfx1250) — matrix operand order repaired and checked by reading, not compiled, not run on hardware
//
#include <hip/hip_runtime.h>
#include <hip/hip_bf16.h>
#include <cstdint>

#define D_DIM 512
#define E_DIM 64
#define KNN 8
#define EPSV 1e-8f
#define TAUV 1.5f

// LDS layout (dynamic shared):
//   [0, 64KB)   : B double buffer (2 x 32KB, 32 rows x 512 halves each)
//   [64KB,80KB) : C staging (8 waves x 16 rows x 32 cols x f32)
#define B_BUF_BYTES 32768
#define TILEC_OFF   (2 * B_BUF_BYTES)
#define SMEM_BYTES  (2 * B_BUF_BYTES + 8 * 16 * 32 * 4)

typedef _Float16 v16h __attribute__((ext_vector_type(16)));
typedef _Float16 v8h  __attribute__((ext_vector_type(8)));
typedef float    v8f  __attribute__((ext_vector_type(8)));
typedef unsigned int u32x4 __attribute__((ext_vector_type(4)));
typedef int i32x4 __attribute__((ext_vector_type(4)));
typedef int i32x8 __attribute__((ext_vector_type(8)));

#if __has_builtin(__builtin_amdgcn_tensor_load_to_lds) && \
    __has_builtin(__builtin_amdgcn_s_wait_tensorcnt)
#define USE_TDM 1
#else
#define USE_TDM 0
#endif

#if USE_TDM
// ---------------------------------------------------------------------------
// TDM: DMA one contiguous 32KB B-tile (32 rows x 512 halves) into LDS.
// D# packed per CDNA5 ISA ch.8: 1-D tile of 16384 2-byte elements.
// ---------------------------------------------------------------------------
__device__ __forceinline__ void tdm_load_tile(const _Float16* gsrc,
                                              unsigned lds_byte_off) {
  const unsigned long long ga = (unsigned long long)(uintptr_t)gsrc;
  u32x4 g0;
  g0[0] = 1u;                                    // count=1 (valid), user mode
  g0[1] = lds_byte_off;                          // lds_addr (bytes)
  g0[2] = (unsigned)(ga & 0xFFFFFFFFu);          // global_addr[31:0]
  g0[3] = (unsigned)((ga >> 32) & 0x01FFFFFFu)   // global_addr[56:32]
        | (2u << 30);                            // type = 2 ("image")
  i32x8 g1;
  g1[0] = (int)(1u << 16);         // wg_mask=0, data_size=1 (2 bytes)
  g1[1] = (int)(16384u << 16);     // tensor_dim0[15:0] = 16384
  g1[2] = (int)(1u << 16);         // tensor_dim0[31:16]=0 | tensor_dim1=1
  g1[3] = (int)(16384u << 16);     // tile_dim0 = 16384 elements
  g1[4] = 1;                       // tile_dim1 = 1, tile_dim2 = 0
  g1[5] = 16384;                   // tensor_dim0_stride
  g1[6] = 0;
  g1[7] = 0;
  i32x4 gz = {0, 0, 0, 0};
#if defined(__clang_major__) && (__clang_major__ >= 23)
  i32x8 gz8 = {0, 0, 0, 0, 0, 0, 0, 0};
  __builtin_amdgcn_tensor_load_to_lds(g0, g1, gz, gz, gz8, 0);
#else
  __builtin_amdgcn_tensor_load_to_lds(g0, g1, gz, gz, 0);
#endif
}
#else
// Fallback: cooperative 32KB copy (tile rows are contiguous in memory).
__device__ __forceinline__ void coop_copy_tile(const _Float16* gsrc,
                                               char* smem, int buf, int tid) {
  const uint4* s = (const uint4*)gsrc;
  uint4* d = (uint4*)(smem + buf * B_BUF_BYTES);
#pragma unroll
  for (int i = 0; i < 8; ++i) d[tid + 256 * i] = s[tid + 256 * i];
}
#endif

// ---------------------------------------------------------------------------
// Sorted (descending) top-8 insertion, register resident after unroll.
// Strict '>' keeps earlier (lower-index) candidates ahead on ties.
// ---------------------------------------------------------------------------
__device__ __forceinline__ void topk_insert(float (&tv)[KNN], int (&ti)[KNN],
                                            float v, int idx) {
  if (v > tv[KNN - 1]) {
    tv[KNN - 1] = v;
    ti[KNN - 1] = idx;
#pragma unroll
    for (int s = KNN - 1; s >= 1; --s) {
      if (tv[s] > tv[s - 1]) {
        float tf = tv[s]; tv[s] = tv[s - 1]; tv[s - 1] = tf;
        int   tn = ti[s]; ti[s] = ti[s - 1]; ti[s - 1] = tn;
      }
    }
  }
}

#define WMMA_F16(A, B, C) \
  __builtin_amdgcn_wmma_f32_16x16x32_f16(false, (A), false, (B), (short)0, (C), \
                                         false, false)

// ---------------------------------------------------------------------------
// One 32-wide column tile: 2 sub-tiles x 16 K-chunks as FOUR independent
// accumulator chains (K split low/high per sub-tile), plus scheduler-group
// directives pinning the stream to  [16 ds_read]([4 wmma][8 ds_read])x6
// [8 wmma]  so DS reads run one full group ahead of their consuming WMMAs
// (partial DScnt waits instead of full drains).
// ---------------------------------------------------------------------------
__device__ __forceinline__ void gemm_tile32(const _Float16* b0,
                                            const _Float16* b1,
                                            const v16h (&afr)[16],
                                            v8f& out0, v8f& out1) {
  v8f a00 = {0.f, 0.f, 0.f, 0.f, 0.f, 0.f, 0.f, 0.f};
  v8f a01 = a00, a10 = a00, a11 = a00;
  v16h p0 = *(const v16h*)(b0);            // sub-tile0, K-chunk 0
  v16h p8 = *(const v16h*)(b0 + 8 * 32);   // sub-tile0, K-chunk 8
  v16h q0 = *(const v16h*)(b1);            // sub-tile1, K-chunk 0
  v16h q8 = *(const v16h*)(b1 + 8 * 32);   // sub-tile1, K-chunk 8
#pragma unroll
  for (int i = 0; i < 8; ++i) {
    a00 = WMMA_F16(afr[i],     p0, a00);
    a10 = WMMA_F16(afr[i],     q0, a10);
    a01 = WMMA_F16(afr[i + 8], p8, a01);
    a11 = WMMA_F16(afr[i + 8], q8, a11);
    if (i < 7) {
      p0 = *(const v16h*)(b0 + (i + 1) * 32);
      p8 = *(const v16h*)(b0 + (i + 9) * 32);
      q0 = *(const v16h*)(b1 + (i + 1) * 32);
      q8 = *(const v16h*)(b1 + (i + 9) * 32);
    }
  }
#if __has_builtin(__builtin_amdgcn_sched_group_barrier)
  // 64 ds_load_b128 + 32 WMMA in this region.  Masks: 0x100=DS-read,
  // 0x008=MFMA/WMMA.  Loads stay one 8-load group ahead of their WMMAs.
  __builtin_amdgcn_sched_group_barrier(0x100, 16, 0);
#pragma unroll
  for (int g = 0; g < 6; ++g) {
    __builtin_amdgcn_sched_group_barrier(0x008, 4, 0);
    __builtin_amdgcn_sched_group_barrier(0x100, 8, 0);
  }
  __builtin_amdgcn_sched_group_barrier(0x008, 8, 0);
#endif
  out0 = a00 + a01;
  out1 = a10 + a11;
}

// ---------------------------------------------------------------------------
// Kernel 1: row-normalize z (f32) -> z_norm (f16). One block per row.
// ---------------------------------------------------------------------------
__global__ __launch_bounds__(256) void normalize_kernel(
    const float* __restrict__ z, _Float16* __restrict__ zn) {
  __shared__ float red[256];
  const int row = blockIdx.x;
  const int t = threadIdx.x;
  const float* zr = z + (size_t)row * D_DIM;
  float a = zr[t];
  float b = zr[t + 256];
  red[t] = a * a + b * b;
  __syncthreads();
#pragma unroll
  for (int s = 128; s > 0; s >>= 1) {
    if (t < s) red[t] += red[t + s];
    __syncthreads();
  }
  const float inv = 1.0f / fmaxf(sqrtf(red[0]), 1e-12f);
  _Float16* znr = zn + (size_t)row * D_DIM;
  znr[t]       = (_Float16)(a * inv);
  znr[t + 256] = (_Float16)(b * inv);
}

// ---------------------------------------------------------------------------
// Kernel 2: fused sim = Zn*Zn^T (f16 WMMA, f32 acc) + per-row top-8 + JS
// coherence.  Block = 128 rows (8 waves x 16 rows).  All waves sweep the same
// 32-wide column tile; the 32KB B-tile is TDM-DMA'd into double-buffered LDS
// once per block.  Lanes 0-15 / 16-31 keep top-8 for the low/high 16 columns
// of the same rows; merged at the end with 16 shuffles.
// ---------------------------------------------------------------------------
__global__ __launch_bounds__(256, 1) void sim_topk_coh_kernel(
    const _Float16* __restrict__ zn, const float* __restrict__ gates,
    float* __restrict__ coh_row, int N) {
  extern __shared__ char smem[];
  float* tileC = (float*)(smem + TILEC_OFF);  // 8 waves x 16 x 32 f32

  const int tid  = threadIdx.x;
  const int w    = tid >> 5;
  const int lane = tid & 31;
  const int m16  = lane & 15;
  const int hi   = lane >> 4;
  const int row0 = blockIdx.x * 128 + w * 16;   // this wave's 16 rows
  const int NT2  = N >> 5;                      // 32-wide column tiles

  // ---- A fragments: 16 rows x 512 K as 16 chunks of K=32 (128 VGPRs) ----
  // A 16x32 layout (matches B): lanes 0-15 hold K=0..15,
  // lanes 16-31 hold K=16..31, M = lane&15.
  v16h afr[16];
  const _Float16* arow = zn + (size_t)(row0 + m16) * D_DIM;
#pragma unroll
  for (int ck = 0; ck < 16; ++ck) {
    const int kb = ck * 32 + hi * 16;
    v8h lo = *(const v8h*)(arow + kb);
    v8h hh = *(const v8h*)(arow + kb + 8);
    v16h a;
#pragma unroll
    for (int i = 0; i < 8; ++i) { a[i] = lo[i]; a[i + 8] = hh[i]; }
    afr[ck] = a;
  }

  // ---- Prologue: stage tile 0 into buffer 0 ----
#if USE_TDM
  if (w == 0) {
    tdm_load_tile(zn, 0u);
    __builtin_amdgcn_s_wait_tensorcnt(0);
  }
#else
  coop_copy_tile(zn, smem, 0, tid);
#endif
  __syncthreads();

  float tv[KNN]; int ti[KNN];
#pragma unroll
  for (int s = 0; s < KNN; ++s) { tv[s] = -3.0e38f; ti[s] = 0; }

  for (int c = 0; c < NT2; ++c) {
    const int buf = c & 1;
    const int colbase = c * 32;

    // Kick DMA of the next 32KB tile into the other buffer (overlaps compute).
#if USE_TDM
    if (w == 0 && c + 1 < NT2)
      tdm_load_tile(zn + (size_t)(colbase + 32) * D_DIM,
                    (unsigned)((buf ^ 1) * B_BUF_BYTES));
#else
    if (c + 1 < NT2)
      coop_copy_tile(zn + (size_t)(colbase + 32) * D_DIM, smem, buf ^ 1, tid);
#endif

    // B 32x16 fragment base in LDS: N = lane&15, lanes 0-15 hold K=0..15,
    // lanes 16-31 hold K=16..31 -> one contiguous 16-half read per lane.
    const _Float16* bbase =
        (const _Float16*)(smem + buf * B_BUF_BYTES) + m16 * D_DIM + hi * 16;

    v8f acc0, acc1;
    gemm_tile32(bbase, bbase + 16 * D_DIM, afr, acc0, acc1);

    // Stage C (lane/VGPR j holds M = j + 8*hi, N = lane&15).
    float* tc = tileC + w * 512;   // 16 rows x 32 cols
#pragma unroll
    for (int j = 0; j < 8; ++j) {
      tc[(j + 8 * hi) * 32 + m16]      = acc0[j];
      tc[(j + 8 * hi) * 32 + 16 + m16] = acc1[j];
    }
    __builtin_amdgcn_wave_barrier();
    asm volatile("" ::: "memory");   // same-wave DS ops are in-order in HW

    // All 32 lanes scan: lane = (colgroup hi, row m16).
    {
      const int row = row0 + m16;
#pragma unroll
      for (int n = 0; n < 16; ++n) {
        const int col = colbase + hi * 16 + n;
        float v = tc[m16 * 32 + hi * 16 + n];
        if (col == row) v = 0.0f;    // reference zeroes the diagonal
        topk_insert(tv, ti, v, col);
      }
    }
    asm volatile("" ::: "memory");
    __builtin_amdgcn_wave_barrier();

#if USE_TDM
    if (w == 0) __builtin_amdgcn_s_wait_tensorcnt(0);  // next tile landed
#endif
    __syncthreads();   // publish next buffer / retire current one
  }

  // ---- Merge high-column shadow lists (lanes 16-31) into lanes 0-15 ----
#pragma unroll
  for (int s = 0; s < KNN; ++s) {
    float ov = __shfl(tv[s], lane + 16, 32);
    int   oi = __shfl(ti[s], lane + 16, 32);
    if (lane < 16) topk_insert(tv, ti, ov, oi);
  }

  // ---- JS-weighted coherence: each wave finalizes its own 16 rows ----
  if (lane < 16) {
    const int row = row0 + lane;
    const float* gi = gates + (size_t)row * E_DIM;
    float csum = 0.f;
    for (int s = 0; s < KNN; ++s) {
      const float* gj = gates + (size_t)ti[s] * E_DIM;
      float klp = 0.f, klq = 0.f;
      for (int e = 0; e < E_DIM; ++e) {
        float p = fmaxf(gi[e], EPSV);
        float q = fmaxf(gj[e], EPSV);
        float mm = 0.5f * (p + q);
        klp += p * __logf(p / mm);
        klq += q * __logf(q / mm);
      }
      csum += tv[s] * 0.5f * (klp + klq);
    }
    coh_row[row] = csum;
  }
}

// ---------------------------------------------------------------------------
// Kernel 3: expert histogram of topk_idx[:,0] + L_sep (deterministic).
// ---------------------------------------------------------------------------
__global__ __launch_bounds__(64) void hist_sep_kernel(
    const int* __restrict__ topk, float* __restrict__ sep, int N) {
  __shared__ float cf[E_DIM];
  const int e = threadIdx.x;
  int cnt = 0;
  for (int n = 0; n < N; ++n) cnt += (topk[2 * n] == e) ? 1 : 0;
  cf[e] = (float)cnt;
  __syncthreads();
  if (e == 0) {
    float s = 0.f;
    for (int i = 0; i < E_DIM; ++i) s += cf[i];
    const float nbar = fmaxf(s / (float)E_DIM, EPSV);
    float acc = 0.f;
    for (int i = 0; i < E_DIM; ++i) {
      float o = fmaxf(cf[i] / nbar - TAUV, 0.f);
      acc += o * o;
    }
    sep[0] = acc / (float)E_DIM;
  }
}

// ---------------------------------------------------------------------------
// Kernel 4: g_bar = column mean of gates (fixed-order, deterministic).
// ---------------------------------------------------------------------------
__global__ __launch_bounds__(64) void gbar_kernel(
    const float* __restrict__ gates, float* __restrict__ gbar, int N) {
  const int e = threadIdx.x;
  float s = 0.f;
  for (int n = 0; n < N; ++n) s += gates[(size_t)n * E_DIM + e];
  gbar[e] = s / (float)N;
}

// ---------------------------------------------------------------------------
// Kernel 5: per-row JS(gates_row, g_bar).
// ---------------------------------------------------------------------------
__global__ __launch_bounds__(256) void ali_kernel(
    const float* __restrict__ gates, const float* __restrict__ gbar,
    float* __restrict__ ali, int N) {
  const int row = blockIdx.x * blockDim.x + threadIdx.x;
  if (row >= N) return;
  const float* gi = gates + (size_t)row * E_DIM;
  float klp = 0.f, klq = 0.f;
  for (int e = 0; e < E_DIM; ++e) {
    float p = fmaxf(gi[e], EPSV);
    float q = fmaxf(gbar[e], EPSV);
    float mm = 0.5f * (p + q);
    klp += p * __logf(p / mm);
    klq += q * __logf(q / mm);
  }
  ali[row] = 0.5f * (klp + klq);
}

// ---------------------------------------------------------------------------
// Kernel 6: deterministic final reduction -> [L_coh, L_sep, L_ali, loss].
// ---------------------------------------------------------------------------
__global__ __launch_bounds__(256) void finalize_kernel(
    const float* __restrict__ coh, const float* __restrict__ ali,
    const float* __restrict__ sep, float* __restrict__ out, int N) {
  __shared__ float r1[256];
  __shared__ float r2[256];
  const int t = threadIdx.x;
  float a = 0.f, b = 0.f;
  for (int i = t; i < N; i += 256) { a += coh[i]; b += ali[i]; }
  r1[t] = a; r2[t] = b;
  __syncthreads();
#pragma unroll
  for (int s = 128; s > 0; s >>= 1) {
    if (t < s) { r1[t] += r1[t + s]; r2[t] += r2[t + s]; }
    __syncthreads();
  }
  if (t == 0) {
    const float Lcoh = r1[0] / ((float)N * (float)KNN);
    const float Lali = r2[0] / (float)N;
    const float Lsep = sep[0];
    out[0] = Lcoh;
    out[1] = Lsep;
    out[2] = Lali;
    out[3] = 0.1f * Lcoh + 0.05f * Lsep + 0.01f * Lali;
  }
}

// ---------------------------------------------------------------------------
extern "C" void kernel_launch(void* const* d_in, const int* in_sizes, int n_in,
                              void* d_out, int out_size, void* d_ws, size_t ws_size,
                              hipStream_t stream) {
  (void)n_in; (void)out_size; (void)ws_size;
  const float* z     = (const float*)d_in[0];
  const float* gates = (const float*)d_in[1];
  const int*   topk  = (const int*)d_in[2];
  const int N = in_sizes[0] / D_DIM;  // 16384

  char* ws = (char*)d_ws;
  size_t off = 0;
  _Float16* zn  = (_Float16*)(ws + off); off += (size_t)N * D_DIM * sizeof(_Float16);
  float*    coh = (float*)(ws + off);    off += (size_t)N * sizeof(float);
  float*    ali = (float*)(ws + off);    off += (size_t)N * sizeof(float);
  float*    gbr = (float*)(ws + off);    off += E_DIM * sizeof(float);
  float*    sep = (float*)(ws + off);    off += sizeof(float);
  float*    out = (float*)d_out;

  normalize_kernel<<<N, 256, 0, stream>>>(z, zn);
  sim_topk_coh_kernel<<<N / 128, 256, SMEM_BYTES, stream>>>(zn, gates, coh, N);
  hist_sep_kernel<<<1, 64, 0, stream>>>(topk, sep, N);
  gbar_kernel<<<1, 64, 0, stream>>>(gates, gbr, N);
  ali_kernel<<<(N + 255) / 256, 256, 0, stream>>>(gates, gbr, ali, N);
  finalize_kernel<<<1, 256, 0, stream>>>(coh, ali, sep, out, N);
}
